// WindowAttention_16028817949004
// MI455X (gfx1250) — matrix operand order repaired and checked by reading, not compiled, not run on hardware
//
#include <hip/hip_runtime.h>
#include <hip/hip_fp16.h>
#include <hip/hip_bf16.h>

typedef __attribute__((ext_vector_type(16))) _Float16 v16h;
typedef __attribute__((ext_vector_type(8)))  _Float16 v8h;
typedef __attribute__((ext_vector_type(8)))  float    v8f;

#define NTOK 49      // window tokens (7*7)
#define NPAD 64      // padded tokens
#define DIM  384
#define NH   12
#define HD   32      // head dim
#define NWIN 64      // windows per image (mask leading dim)
#define SCALE 0.17677669529663687f  // 1/sqrt(32)

// ---- WMMA fragment helpers (wave32, 16x16x32 f16 -> f32) ----

// A-matrix (16x32, 16-bit), source row-major [m][k]:
// lane<16 holds M=lane, K in {0..7,16..23}; lane>=16 holds M=lane-16,
// K in {8..15,24..31}. Two contiguous 8-half (16B) chunks per lane.
__device__ __forceinline__ v16h a_frag(const _Float16* base, int ld, int lane) {
  const _Float16* row = base + (lane & 15) * ld + ((lane < 16) ? 0 : 8);
  v8h lo = *(const v8h*)(row);
  v8h hi = *(const v8h*)(row + 16);
  v16h a;
#pragma unroll
  for (int i = 0; i < 8; ++i) { a[i] = lo[i]; a[i + 8] = hi[i]; }
  return a;
}

// B-matrix (32x16) from a matrix stored [n][k] row-major (i.e. transposed
// relative to the logical KxN operand). Lane reads two 8-half chunks:
// column n = lane&15, K in (lane<16 ? {0..7,16..23} : {8..15,24..31}). Two b128 loads.
__device__ __forceinline__ v16h bT_frag(const _Float16* base, int ld, int lane) {
  const _Float16* p = base + (lane & 15) * ld + ((lane < 16) ? 0 : 8);
  v8h lo = *(const v8h*)(p);
  v8h hi = *(const v8h*)(p + 16);
  v16h b;
#pragma unroll
  for (int i = 0; i < 8; ++i) { b[i] = lo[i]; b[i + 8] = hi[i]; }
  return b;
}

__device__ __forceinline__ v8f wmma16(v16h a, v16h b, v8f c) {
  return __builtin_amdgcn_wmma_f32_16x16x32_f16(false, a, false, b, (short)0, c,
                                                false, false);
}

// ---- prep: f32 weights -> f16, transposed to [out_col][in_k] ----
__global__ void prep_weights(const float* __restrict__ qkv_w,   // [384,1152]
                             const float* __restrict__ proj_w,  // [384,384]
                             _Float16* __restrict__ qkv_wt,     // [1152,384]
                             _Float16* __restrict__ proj_wt)    // [384,384]
{
  int i = blockIdx.x * 256 + threadIdx.x;
  if (i < 3 * DIM * DIM) {            // 1152*384
    int n = i / DIM, k = i % DIM;
    qkv_wt[i] = (_Float16)qkv_w[(size_t)k * (3 * DIM) + n];
  }
  if (i < DIM * DIM) {                // 384*384
    int n = i / DIM, k = i % DIM;
    proj_wt[i] = (_Float16)proj_w[(size_t)k * DIM + n];
  }
}

__global__ void __launch_bounds__(256, 1)
win_attn_kernel(const float*    __restrict__ x,        // [4096,49,384]
                const float*    __restrict__ mask,     // [64,49,49]
                const _Float16* __restrict__ qkv_wt,   // [1152,384] f16 (n-major)
                const float*    __restrict__ qkv_b,    // [1152]
                const _Float16* __restrict__ proj_wt,  // [384,384] f16 (n-major)
                const float*    __restrict__ proj_b,   // [384]
                const float*    __restrict__ bias_tab, // [169,12]
                const int*      __restrict__ rel_idx,  // [49,49]
                float*          __restrict__ out)      // [4096,49,384]
{
  __shared__ __align__(16) _Float16 Xf[NPAD * DIM];     // 48 KB  x in f16 (padded)
  __shared__ __align__(16) _Float16 Ob[NPAD * DIM];     // 48 KB  attn output (f16)
  __shared__ __align__(16) _Float16 Qh[NPAD * HD];      // 4 KB   Q  [token][chan] (pre-scaled)
  __shared__ __align__(16) _Float16 Kh[NPAD * HD];      // 4 KB   K  [token][chan] (== [n][k] for S)
  __shared__ __align__(16) _Float16 Vt[HD * NPAD];      // 4 KB   V^T [chan][token] (== [n][k] for O)
  __shared__ __align__(16) float    Sf[NPAD * NPAD];    // 16 KB  scores f32
  __shared__ __align__(16) _Float16 Pf[NPAD * NPAD];    // 8 KB   probs f16

  const int tid  = threadIdx.x;
  const int lane = tid & 31;
  const int wid  = tid >> 5;           // 8 waves of 32
  const int b    = blockIdx.x;         // window index 0..4095

  // ---- Stage 0: load x tile -> f16 LDS, zero-pad rows 49..63 ----
  {
    const float* xb = x + (size_t)b * NTOK * DIM;
    for (int i = tid; i < NTOK * (DIM / 4); i += 256) {
      int r = i / (DIM / 4), c4 = i % (DIM / 4);
      float4 v = ((const float4*)(xb + (size_t)r * DIM))[c4];
      _Float16* d = &Xf[r * DIM + c4 * 4];
      d[0] = (_Float16)v.x; d[1] = (_Float16)v.y;
      d[2] = (_Float16)v.z; d[3] = (_Float16)v.w;
    }
    for (int i = tid; i < (NPAD - NTOK) * DIM; i += 256)
      Xf[NTOK * DIM + i] = (_Float16)0.0f;
  }

  const float* maskw = mask + (size_t)(b % NWIN) * NTOK * NTOK;

#pragma unroll 1
  for (int h = 0; h < NH; ++h) {
    __syncthreads();  // Xf ready / previous head done with Qh,Kh,Vt,Pf

    // ---- Stage 1: per-head QKV GEMM: [64,384] x [384,32] (x3) ----
    // Wave handles one 16x16 (m0,n0) tile for ALL of q,k,v: one A-frag
    // feeds 3 WMMAs per K-step (3x fewer LDS A loads).
    {
      int m0 = (wid >> 1) * 16, n0 = (wid & 1) * 16;
      const _Float16* wq = qkv_wt + (size_t)(0 * DIM + h * HD + n0) * DIM;
      const _Float16* wk = qkv_wt + (size_t)(1 * DIM + h * HD + n0) * DIM;
      const _Float16* wv = qkv_wt + (size_t)(2 * DIM + h * HD + n0) * DIM;
      v8f cq = {}, ck = {}, cv = {};
#pragma unroll
      for (int ki = 0; ki < DIM / 32; ++ki) {
        v16h a = a_frag(&Xf[m0 * DIM + ki * 32], DIM, lane);
        cq = wmma16(a, bT_frag(wq + ki * 32, DIM, lane), cq);
        ck = wmma16(a, bT_frag(wk + ki * 32, DIM, lane), ck);
        cv = wmma16(a, bT_frag(wv + ki * 32, DIM, lane), cv);
      }
      int n   = lane & 15;
      int col = n0 + n;                               // 0..31 within head
      float bq = qkv_b[0 * DIM + h * HD + col];
      float bk = qkv_b[1 * DIM + h * HD + col];
      float bv = qkv_b[2 * DIM + h * HD + col];
#pragma unroll
      for (int v = 0; v < 8; ++v) {
        int m = m0 + ((lane < 16) ? v : v + 8);
        Qh[m * HD + col]   = (_Float16)((cq[v] + bq) * SCALE);  // fold 1/sqrt(hd)
        Kh[m * HD + col]   = (_Float16)(ck[v] + bk);            // natural [t][c]
        Vt[col * NPAD + m] = (_Float16)(cv[v] + bv);            // transposed [c][t]
      }
    }
    __syncthreads();

    // ---- Stage 2: S = Q * K^T  (16 tiles of 16x16, single K=32 step) ----
    // B operand [k=chan][n=token] is Kh stored [token][chan] = [n][k]: contiguous.
#pragma unroll 1
    for (int t2 = wid * 2; t2 < wid * 2 + 2; ++t2) {
      int mt = t2 >> 2, nt = t2 & 3;
      v16h a  = a_frag(&Qh[mt * 16 * HD], HD, lane);
      v16h bb = bT_frag(&Kh[nt * 16 * HD], HD, lane);
      v8f c = {};
      c = wmma16(a, bb, c);
      int n = nt * 16 + (lane & 15);
#pragma unroll
      for (int v = 0; v < 8; ++v) {
        int m = mt * 16 + ((lane < 16) ? v : v + 8);
        Sf[m * NPAD + n] = c[v];
      }
    }
    __syncthreads();

    // ---- Stage 3: bias + mask + row softmax -> Pf (f16) ----
    if (tid < NPAD) {
      int r = tid;
      if (r >= NTOK) {
#pragma unroll 1
        for (int kk = 0; kk < NPAD; ++kk) Pf[r * NPAD + kk] = (_Float16)0.0f;
      } else {
        float mx = -1e30f;
#pragma unroll 1
        for (int kk = 0; kk < NTOK; ++kk) {
          float s = Sf[r * NPAD + kk]
                  + bias_tab[rel_idx[r * NTOK + kk] * NH + h]
                  + maskw[r * NTOK + kk];
          Sf[r * NPAD + kk] = s;
          mx = fmaxf(mx, s);
        }
        float sum = 0.0f;
#pragma unroll 1
        for (int kk = 0; kk < NTOK; ++kk) {
          float e = __expf(Sf[r * NPAD + kk] - mx);
          Sf[r * NPAD + kk] = e;
          sum += e;
        }
        float inv = 1.0f / sum;
#pragma unroll 1
        for (int kk = 0; kk < NTOK; ++kk)
          Pf[r * NPAD + kk] = (_Float16)(Sf[r * NPAD + kk] * inv);
        for (int kk = NTOK; kk < NPAD; ++kk) Pf[r * NPAD + kk] = (_Float16)0.0f;
      }
    }
    __syncthreads();

    // ---- Stage 4: O_h = P * V  (8 tiles: 4 Mtiles x 2 Ntiles, K=64) ----
    // B operand [k=token][n=chan] is Vt stored [chan][token] = [n][k]: contiguous.
    {
      int mt = wid >> 1, nt = wid & 1;
      v8f c = {};
#pragma unroll
      for (int ki = 0; ki < 2; ++ki) {
        v16h a  = a_frag(&Pf[mt * 16 * NPAD + ki * 32], NPAD, lane);
        v16h bb = bT_frag(&Vt[nt * 16 * NPAD + ki * 32], NPAD, lane);
        c = wmma16(a, bb, c);
      }
      int n = nt * 16 + (lane & 15);
#pragma unroll
      for (int v = 0; v < 8; ++v) {
        int m = mt * 16 + ((lane < 16) ? v : v + 8);
        Ob[m * DIM + h * HD + n] = (_Float16)c[v];
      }
    }
  }
  __syncthreads();

  // ---- Stage 5: out = O * proj_w + proj_b ----
  // 4 Mtiles x 24 Ntiles; wave takes Ntiles {wid, wid+8, wid+16} so one
  // A-frag feeds 3 WMMAs per K-step.
#pragma unroll 1
  for (int mt = 0; mt < 4; ++mt) {
    int m0 = mt * 16;
    const _Float16* w0 = proj_wt + (size_t)((wid +  0) * 16) * DIM;
    const _Float16* w1 = proj_wt + (size_t)((wid +  8) * 16) * DIM;
    const _Float16* w2 = proj_wt + (size_t)((wid + 16) * 16) * DIM;
    v8f c0 = {}, c1 = {}, c2 = {};
#pragma unroll
    for (int ki = 0; ki < DIM / 32; ++ki) {
      v16h a = a_frag(&Ob[m0 * DIM + ki * 32], DIM, lane);
      c0 = wmma16(a, bT_frag(w0 + ki * 32, DIM, lane), c0);
      c1 = wmma16(a, bT_frag(w1 + ki * 32, DIM, lane), c1);
      c2 = wmma16(a, bT_frag(w2 + ki * 32, DIM, lane), c2);
    }
    int n = lane & 15;
#pragma unroll
    for (int j = 0; j < 3; ++j) {
      int n0  = (wid + 8 * j) * 16;
      int col = n0 + n;
      float pb = proj_b[col];
      v8f c = (j == 0) ? c0 : (j == 1) ? c1 : c2;
#pragma unroll
      for (int v = 0; v < 8; ++v) {
        int m = m0 + ((lane < 16) ? v : v + 8);
        if (m < NTOK)
          out[((size_t)b * NTOK + m) * DIM + col] = c[v] + pb;
      }
    }
  }
}

extern "C" void kernel_launch(void* const* d_in, const int* in_sizes, int n_in,
                              void* d_out, int out_size, void* d_ws, size_t ws_size,
                              hipStream_t stream) {
  const float* x        = (const float*)d_in[0];
  const float* mask     = (const float*)d_in[1];
  const float* qkv_w    = (const float*)d_in[2];
  const float* qkv_b    = (const float*)d_in[3];
  const float* proj_w   = (const float*)d_in[4];
  const float* proj_b   = (const float*)d_in[5];
  const float* bias_tab = (const float*)d_in[6];
  const int*   rel_idx  = (const int*)d_in[7];
  float* out = (float*)d_out;
  (void)in_sizes; (void)n_in; (void)out_size; (void)ws_size;

  // f16 transposed weights in workspace: qkv 1152*384*2 = 884736 B, proj 294912 B
  _Float16* qkv_wt  = (_Float16*)d_ws;
  _Float16* proj_wt = (_Float16*)((char*)d_ws + (size_t)3 * DIM * DIM * 2);

  int prep_elems = 3 * DIM * DIM;                       // max of the two
  prep_weights<<<(prep_elems + 255) / 256, 256, 0, stream>>>(qkv_w, proj_w,
                                                            qkv_wt, proj_wt);
  win_attn_kernel<<<4096, 256, 0, stream>>>(x, mask, qkv_wt, qkv_b, proj_wt,
                                            proj_b, bias_tab, rel_idx, out);
}
